// StyleLattice_48619029791167
// MI455X (gfx1250) — matrix operand order repaired and checked by reading, not compiled, not run on hardware
//
#include <hip/hip_runtime.h>

// ---------------------------------------------------------------------------
// CDNA5 (gfx1250, wave32) implementation of the StyleLattice reference.
// All GEMMs run on v_wmma_f32_16x16x32_f16 (f16 inputs, f32 accumulate).
// Group-encoder layer 2 is linear before the mean over S, so the mean is
// taken first (16x fewer layer-2 flops). grp_feats streaming (268 MB) sets
// the roofline floor (~13 us @ 23.3 TB/s); tile loads are double-buffered in
// VGPRs so HBM latency hides behind WMMA compute.
// ---------------------------------------------------------------------------

typedef __attribute__((ext_vector_type(16))) _Float16 v16h;
typedef __attribute__((ext_vector_type(2)))  _Float16 v2h;
typedef __attribute__((ext_vector_type(8)))  float    v8f;

#define WAVES    8
#define THREADS  (WAVES * 32)
#define NBLOCKS  256

// ---- shared memory layout (byte offsets, union of phase1/phase2 weights) ----
#define O_W1T    0            // [128][64]  f16  16384 B  (ei_w1 / eg_w1 transposed)
#define O_W2T    16384        // [128][128] f16  32768 B  (ei_w2 / eg_w2 transposed)
#define O_MUWT   49152        // [16][128]  f16   4096 B
#define O_LVWT   53248        // [16][128]  f16   4096 B
#define O_CTXWT  57344        // [128][32]  f16   8192 B  (phase2)
#define O_PRMU   65536        // [16][32]   f16   1024 B  (phase2)
#define O_PRLV   66560        // [16][32]   f16   1024 B  (phase2)
#define O_CB     67584        // [64][16]   f32   4096 B  (cb_i / cb_g)
#define O_B1     71680        // [128] f32
#define O_B2     72192        // [128] f32
#define O_CTXB   72704        // [128] f32
#define O_MUB    73216        // [16]  f32
#define O_LVB    73280
#define O_PMUB   73344
#define O_PLVB   73408
#define O_STAGE  73472        // per wave: xstage 2048 B (f16 16x64) + hstage 4096 B (f16 16x128)
#define STAGE_PER_WAVE 6144
#define SMEM_TOTAL (O_STAGE + WAVES * STAGE_PER_WAVE)   // 122624 B (< 320 KB WGP LDS)

struct Params {
    const float *ind, *grp, *ctx, *eps_i, *eps_g;
    const float *ei_w1, *ei_b1, *ei_w2, *ei_b2, *ei_mu_w, *ei_mu_b, *ei_lv_w, *ei_lv_b, *cb_i;
    const float *eg_w1, *eg_b1, *eg_w2, *eg_b2, *eg_ctx_w, *eg_ctx_b;
    const float *eg_mu_w, *eg_mu_b, *eg_lv_w, *eg_lv_b, *cb_g;
    const float *pr_mu_w, *pr_mu_b, *pr_lv_w, *pr_lv_b;
    float *out;
    float *ws;
    int B;
};

// CDNA5 split-counter wait for same-wave LDS RAW/WAR through staging buffers.
__device__ __forceinline__ void lds_fence() {
    asm volatile("s_wait_dscnt 0" ::: "memory");
}

__device__ __forceinline__ v8f zero8() {
    v8f z;
#pragma unroll
    for (int i = 0; i < 8; ++i) z[i] = 0.0f;
    return z;
}

__device__ __forceinline__ v8f wmma_f16(v16h a, v16h b, v8f c) {
    // 8 args: (neg_a, A, neg_b, B, c_mod, C, reuse_a, reuse_b)
    return __builtin_amdgcn_wmma_f32_16x16x32_f16(false, a, false, b, (short)0, c, false, false);
}

// A operand (16xK row-major f16 in LDS), one 16x32 K-chunk starting at kc.
// ISA A layout: lane m=lane&15; K(j) = j + 8*(j>=8) + 8*(lane>=16).
__device__ __forceinline__ v16h load_a_lds(const _Float16* X, int ldk, int kc, int lane) {
    const int m   = lane & 15;
    const int hi8 = (lane >> 4) << 3;  // 0 or 8
    const _Float16* p = X + m * ldk + kc + hi8;
    v16h a;
#pragma unroll
    for (int j = 0; j < 8; ++j)  a[j] = p[j];
#pragma unroll
    for (int j = 8; j < 16; ++j) a[j] = p[j + 8];
    return a;
}

// B operand from transposed weights Wt[n][k] (row length ldk halfs).
// ISA B layout: N = nbase + (lane&15); K(j) = kc + j + 8*(j>=8) + 8*(lane>=16) (same as A).
__device__ __forceinline__ v16h load_b_lds(const _Float16* Wt, int ldk, int nbase, int kc, int lane) {
    const int n    = nbase + (lane & 15);
    const int hi16 = ((lane >> 4) & 1) << 3;
    const _Float16* p = Wt + n * ldk + kc + hi16;
    v16h b;
#pragma unroll
    for (int j = 0; j < 8; ++j) b[j] = p[j];
#pragma unroll
    for (int j = 8; j < 16; ++j) b[j] = p[j + 8];
    return b;
}

__device__ __forceinline__ float wave_sum(float v) {
#pragma unroll
    for (int m = 16; m > 0; m >>= 1) v += __shfl_xor(v, m, 32);
    return v;
}

// Global W[K][N] f32  ->  LDS Wt[N][K] f16 (coalesced global reads).
__device__ __forceinline__ void preload_wt(_Float16* dst, const float* src, int K, int N,
                                           int tid, int nthreads) {
    for (int i = tid; i < K * N; i += nthreads) {
        int k = i / N, n = i - k * N;
        dst[n * K + k] = (_Float16)src[i];
    }
}

__device__ __forceinline__ void preload_f32(float* dst, const float* src, int n,
                                            int tid, int nthreads) {
    for (int i = tid; i < n; i += nthreads) dst[i] = src[i];
}

// ---- 16x64 f32 tile staging: coalesced float2 loads into regs, packed v2h
// LDS stores. Lane owns column pair (2*lane, 2*lane+1) of all 16 rows. ----
__device__ __forceinline__ void load_tile64_regs(float2* buf, const float* base,
                                                 int rowStride, int lane) {
#pragma unroll
    for (int j = 0; j < 16; ++j)
        buf[j] = *(const float2*)&base[(size_t)j * rowStride + 2 * lane];
}

__device__ __forceinline__ void store_tile64(_Float16* xst, const float2* buf, int lane) {
#pragma unroll
    for (int j = 0; j < 16; ++j) {
        v2h pk;
        pk[0] = (_Float16)buf[j].x;
        pk[1] = (_Float16)buf[j].y;
        *(v2h*)&xst[j * 64 + 2 * lane] = pk;
    }
}

// ctx tile 16x32 f32 -> f16 LDS (float2 loads, packed stores).
__device__ __forceinline__ void stage_ctx(_Float16* xst, const float* ctx, size_t r0, int lane) {
#pragma unroll
    for (int j = 0; j < 8; ++j) {
        int e0 = 64 * j + 2 * lane;
        int r = e0 >> 5, c = e0 & 31;
        float2 v = *(const float2*)&ctx[(r0 + (size_t)r) * 32 + c];
        v2h pk;
        pk[0] = (_Float16)v.x;
        pk[1] = (_Float16)v.y;
        *(v2h*)&xst[r * 32 + c] = pk;
    }
}

// VQ over 16 rows of d=16 in zst. Row = lane&15; the two 16-lane halves scan
// the low/high halves of the codebook, combined with one shfl_xor(16).
// Tie-break keeps the lower code index (argmin-first semantics).
__device__ __forceinline__ float vq_rows(const float* zst, const float* cb, int ncodes,
                                         float* out_z, size_t r0, int lane) {
    const int row  = lane & 15;
    const int half = lane >> 4;
    const int nk   = ncodes >> 1;
    const int k0   = half * nk;
    float zv[16];
#pragma unroll
    for (int d = 0; d < 16; ++d) zv[d] = zst[row * 16 + d];
    float best = 3.4e38f;
    int   bi   = k0;
    for (int k = k0; k < k0 + nk; ++k) {
        float dist = 0.0f;
#pragma unroll
        for (int d = 0; d < 16; ++d) {
            float df = zv[d] - cb[k * 16 + d];
            dist += df * df;
        }
        if (dist < best) { best = dist; bi = k; }
    }
    float obest = __shfl_xor(best, 16, 32);
    bool  win   = half ? (best < obest) : (best <= obest);
    float msum  = 0.0f;
    if (win) {
        msum = best;
        float* o = out_z + (r0 + (size_t)row) * 16;
#pragma unroll
        for (int d = 0; d < 16; ++d) o[d] = cb[bi * 16 + d];
    }
    return msum;
}

__global__ void style_init(float* ws) {
    if (threadIdx.x < 4) ws[threadIdx.x] = 0.0f;
}

__global__ __launch_bounds__(THREADS) void style_main(Params p) {
    __shared__ __align__(32) unsigned char smem[SMEM_TOTAL];

    _Float16* w1t    = (_Float16*)(smem + O_W1T);
    _Float16* w2t    = (_Float16*)(smem + O_W2T);
    _Float16* muwt   = (_Float16*)(smem + O_MUWT);
    _Float16* lvwt   = (_Float16*)(smem + O_LVWT);
    _Float16* ctxwt  = (_Float16*)(smem + O_CTXWT);
    _Float16* prmuwt = (_Float16*)(smem + O_PRMU);
    _Float16* prlvwt = (_Float16*)(smem + O_PRLV);
    float*    cb     = (float*)(smem + O_CB);
    float*    b1     = (float*)(smem + O_B1);
    float*    b2     = (float*)(smem + O_B2);
    float*    ctxb   = (float*)(smem + O_CTXB);
    float*    mub    = (float*)(smem + O_MUB);
    float*    lvb    = (float*)(smem + O_LVB);
    float*    pmub   = (float*)(smem + O_PMUB);
    float*    plvb   = (float*)(smem + O_PLVB);

    const int tid  = threadIdx.x;
    const int lane = tid & 31;
    const int wid  = tid >> 5;
    const int n    = lane & 15;
    const int hi   = (lane >> 4) & 1;

    _Float16* xst = (_Float16*)(smem + O_STAGE + wid * STAGE_PER_WAVE);
    _Float16* hst = (_Float16*)(smem + O_STAGE + wid * STAGE_PER_WAVE + 2048);
    float*    zst = (float*)xst;   // overlays xst after its last use per tile

    const int NT = p.B >> 4;
    float* out_zi  = p.out;
    float* out_zic = p.out + (size_t)p.B * 16;
    float* out_zg  = p.out + (size_t)p.B * 32;
    float* out_zgc = p.out + (size_t)p.B * 48;

    // ================= Phase 1: IndividualEncoder =================
    preload_wt(w1t,  p.ei_w1,   64, 128, tid, THREADS);
    preload_wt(w2t,  p.ei_w2,  128, 128, tid, THREADS);
    preload_wt(muwt, p.ei_mu_w, 128, 16, tid, THREADS);
    preload_wt(lvwt, p.ei_lv_w, 128, 16, tid, THREADS);
    preload_f32(cb,  p.cb_i, 64 * 16, tid, THREADS);
    preload_f32(b1,  p.ei_b1, 128, tid, THREADS);
    preload_f32(b2,  p.ei_b2, 128, tid, THREADS);
    preload_f32(mub, p.ei_mu_b, 16, tid, THREADS);
    preload_f32(lvb, p.ei_lv_b, 16, tid, THREADS);
    __syncthreads();

    float kli_sum = 0.0f, mse_i = 0.0f;
    for (int t = blockIdx.x * WAVES + wid; t < NT; t += gridDim.x * WAVES) {
        const size_t r0 = (size_t)t * 16;

        // X tile -> xstage (f16, 16x64)
        {
            float2 buf[16];
            load_tile64_regs(buf, p.ind + r0 * 64, 64, lane);
            store_tile64(xst, buf, lane);
        }
        lds_fence();

        // h1 = relu(X @ W1 + b1) -> hstage (f16, 16x128)
#pragma unroll
        for (int nt = 0; nt < 8; ++nt) {
            v8f c = zero8();
#pragma unroll
            for (int kt = 0; kt < 2; ++kt)
                c = wmma_f16(load_a_lds(xst, 64, kt * 32, lane),
                             load_b_lds(w1t, 64, nt * 16, kt * 32, lane), c);
            float bb = b1[nt * 16 + n];
#pragma unroll
            for (int i = 0; i < 8; ++i)
                hst[(i + hi * 8) * 128 + nt * 16 + n] = (_Float16)fmaxf(c[i] + bb, 0.0f);
        }
        lds_fence();

        // h2 = relu(h1 @ W2 + b2), buffered in regs then overwrite hstage
        v8f c2[8];
#pragma unroll
        for (int nt = 0; nt < 8; ++nt) {
            v8f c = zero8();
#pragma unroll
            for (int kt = 0; kt < 4; ++kt)
                c = wmma_f16(load_a_lds(hst, 128, kt * 32, lane),
                             load_b_lds(w2t, 128, nt * 16, kt * 32, lane), c);
            float bb = b2[nt * 16 + n];
#pragma unroll
            for (int i = 0; i < 8; ++i) c2[nt][i] = fmaxf(c[i] + bb, 0.0f);
        }
        lds_fence();
#pragma unroll
        for (int nt = 0; nt < 8; ++nt)
#pragma unroll
            for (int i = 0; i < 8; ++i)
                hst[(i + hi * 8) * 128 + nt * 16 + n] = (_Float16)c2[nt][i];
        lds_fence();

        // mu, lv  (K = 128, single 16-wide N tile each)
        v8f cmu = zero8(), clv = zero8();
#pragma unroll
        for (int kt = 0; kt < 4; ++kt) {
            v16h a = load_a_lds(hst, 128, kt * 32, lane);
            cmu = wmma_f16(a, load_b_lds(muwt, 128, 0, kt * 32, lane), cmu);
            clv = wmma_f16(a, load_b_lds(lvwt, 128, 0, kt * 32, lane), clv);
        }
        float mb = mub[n], lb = lvb[n];
#pragma unroll
        for (int i = 0; i < 8; ++i) {
            int    m   = i + hi * 8;
            size_t off = (r0 + (size_t)m) * 16 + n;
            float  mu  = cmu[i] + mb;
            float  lv  = clv[i] + lb;
            float  z   = mu + p.eps_i[off] * __expf(0.5f * lv);
            out_zic[off] = z;
            kli_sum += 1.0f + lv - mu * mu - __expf(lv);
            zst[m * 16 + n] = z;
        }
        lds_fence();
        mse_i += vq_rows(zst, cb, 64, out_zi, r0, lane);
        lds_fence();
    }
    kli_sum = wave_sum(kli_sum);
    mse_i   = wave_sum(mse_i);
    if (lane == 0) { atomicAdd(&p.ws[0], kli_sum); atomicAdd(&p.ws[1], mse_i); }

    // ================= Phase 2: GroupEncoder (reload weights) =================
    __syncthreads();
    preload_wt(w1t,    p.eg_w1,    64, 128, tid, THREADS);
    preload_wt(w2t,    p.eg_w2,   128, 128, tid, THREADS);
    preload_wt(muwt,   p.eg_mu_w, 128,  16, tid, THREADS);
    preload_wt(lvwt,   p.eg_lv_w, 128,  16, tid, THREADS);
    preload_wt(ctxwt,  p.eg_ctx_w, 32, 128, tid, THREADS);
    preload_wt(prmuwt, p.pr_mu_w,  32,  16, tid, THREADS);
    preload_wt(prlvwt, p.pr_lv_w,  32,  16, tid, THREADS);
    preload_f32(cb,   p.cb_g, 32 * 16, tid, THREADS);
    preload_f32(b1,   p.eg_b1, 128, tid, THREADS);
    preload_f32(b2,   p.eg_b2, 128, tid, THREADS);
    preload_f32(ctxb, p.eg_ctx_b, 128, tid, THREADS);
    preload_f32(mub,  p.eg_mu_b, 16, tid, THREADS);
    preload_f32(lvb,  p.eg_lv_b, 16, tid, THREADS);
    preload_f32(pmub, p.pr_mu_b, 16, tid, THREADS);
    preload_f32(plvb, p.pr_lv_b, 16, tid, THREADS);
    __syncthreads();

    float klg_sum = 0.0f, mse_g = 0.0f;
    for (int t = blockIdx.x * WAVES + wid; t < NT; t += gridDim.x * WAVES) {
        const size_t r0 = (size_t)t * 16;
        const float* gbase = p.grp + r0 * (16 * 64);   // rows stride 16*64 floats

        // mean_s relu(x_s @ W1 + b1): layer2 is linear, so mean BEFORE W2.
        // Double-buffer: prefetch tile s+1 into VGPRs while WMMAs for s run.
        v8f acc1[8];
#pragma unroll
        for (int nt = 0; nt < 8; ++nt) acc1[nt] = zero8();

        float2 buf[16];
        load_tile64_regs(buf, gbase, 16 * 64, lane);           // s = 0
        for (int s = 0; s < 16; ++s) {
            store_tile64(xst, buf, lane);
            lds_fence();
            if (s < 15)
                load_tile64_regs(buf, gbase + (s + 1) * 64, 16 * 64, lane);  // overlap
#pragma unroll
            for (int nt = 0; nt < 8; ++nt) {
                v8f c = zero8();
#pragma unroll
                for (int kt = 0; kt < 2; ++kt)
                    c = wmma_f16(load_a_lds(xst, 64, kt * 32, lane),
                                 load_b_lds(w1t, 64, nt * 16, kt * 32, lane), c);
                float bb = b1[nt * 16 + n];
#pragma unroll
                for (int i = 0; i < 8; ++i) acc1[nt][i] += fmaxf(c[i] + bb, 0.0f);
            }
            lds_fence();
        }

        // mean -> hstage (f16)
#pragma unroll
        for (int nt = 0; nt < 8; ++nt)
#pragma unroll
            for (int i = 0; i < 8; ++i)
                hst[(i + hi * 8) * 128 + nt * 16 + n] = (_Float16)(acc1[nt][i] * 0.0625f);

        // ctx tile -> xstage (f16, 16x32)
        stage_ctx(xst, p.ctx, r0, lane);
        lds_fence();

        // hG = mean @ W2 + b2 + ctx @ Wc + bc
        v8f c2[8];
#pragma unroll
        for (int nt = 0; nt < 8; ++nt) {
            v8f c = zero8();
#pragma unroll
            for (int kt = 0; kt < 4; ++kt)
                c = wmma_f16(load_a_lds(hst, 128, kt * 32, lane),
                             load_b_lds(w2t, 128, nt * 16, kt * 32, lane), c);
            c = wmma_f16(load_a_lds(xst, 32, 0, lane),
                         load_b_lds(ctxwt, 32, nt * 16, 0, lane), c);
            float bb = b2[nt * 16 + n] + ctxb[nt * 16 + n];
#pragma unroll
            for (int i = 0; i < 8; ++i) c2[nt][i] = c[i] + bb;
        }
        lds_fence();
#pragma unroll
        for (int nt = 0; nt < 8; ++nt)
#pragma unroll
            for (int i = 0; i < 8; ++i)
                hst[(i + hi * 8) * 128 + nt * 16 + n] = (_Float16)c2[nt][i];
        lds_fence();

        // mu_g, lv_g from hG; prior mu/lv from ctx
        v8f cmu = zero8(), clv = zero8();
#pragma unroll
        for (int kt = 0; kt < 4; ++kt) {
            v16h a = load_a_lds(hst, 128, kt * 32, lane);
            cmu = wmma_f16(a, load_b_lds(muwt, 128, 0, kt * 32, lane), cmu);
            clv = wmma_f16(a, load_b_lds(lvwt, 128, 0, kt * 32, lane), clv);
        }
        v16h actx = load_a_lds(xst, 32, 0, lane);
        v8f  cpm  = wmma_f16(actx, load_b_lds(prmuwt, 32, 0, 0, lane), zero8());
        v8f  cpl  = wmma_f16(actx, load_b_lds(prlvwt, 32, 0, 0, lane), zero8());

        float mb = mub[n], lb = lvb[n], pmb = pmub[n], plb = plvb[n];
#pragma unroll
        for (int i = 0; i < 8; ++i) {
            int    m   = i + hi * 8;
            size_t off = (r0 + (size_t)m) * 16 + n;
            float  mu  = cmu[i] + mb;
            float  lv  = clv[i] + lb;
            float  pm  = cpm[i] + pmb;
            float  pl  = cpl[i] + plb;
            float  z   = mu + p.eps_g[off] * __expf(0.5f * lv);
            out_zgc[off] = z;
            float d = mu - pm;
            klg_sum += pl - lv + (__expf(lv) + d * d) * __expf(-pl) - 1.0f;
            zst[m * 16 + n] = z;   // overwrites ctx staging (done with it)
        }
        lds_fence();
        mse_g += vq_rows(zst, cb, 32, out_zg, r0, lane);
        lds_fence();
    }
    klg_sum = wave_sum(klg_sum);
    mse_g   = wave_sum(mse_g);
    if (lane == 0) { atomicAdd(&p.ws[2], klg_sum); atomicAdd(&p.ws[3], mse_g); }
}

__global__ void style_finalize(const float* ws, float* out, int B) {
    if (threadIdx.x == 0) {
        float inv = 1.0f / (float)(B * 16);
        float kli = -0.5f * ws[0] * inv;
        float klg =  0.5f * ws[2] * inv;
        float vqi =  0.5f * ws[1] * inv;   // beta_vq * (commit + code) = 0.25 * 2 * mse
        float vqg =  0.5f * ws[3] * inv;
        float* sc = out + (size_t)B * 64;
        sc[0] = 2.0f * (kli + klg) + vqi + vqg;  // loss_style (beta = 2.0)
        sc[1] = kli;
        sc[2] = klg;
    }
}

extern "C" void kernel_launch(void* const* d_in, const int* in_sizes, int n_in,
                              void* d_out, int out_size, void* d_ws, size_t ws_size,
                              hipStream_t stream) {
    Params p;
    p.ind     = (const float*)d_in[0];
    p.grp     = (const float*)d_in[1];
    p.ctx     = (const float*)d_in[2];
    p.eps_i   = (const float*)d_in[3];
    p.eps_g   = (const float*)d_in[4];
    p.ei_w1   = (const float*)d_in[5];
    p.ei_b1   = (const float*)d_in[6];
    p.ei_w2   = (const float*)d_in[7];
    p.ei_b2   = (const float*)d_in[8];
    p.ei_mu_w = (const float*)d_in[9];
    p.ei_mu_b = (const float*)d_in[10];
    p.ei_lv_w = (const float*)d_in[11];
    p.ei_lv_b = (const float*)d_in[12];
    p.cb_i    = (const float*)d_in[13];
    p.eg_w1   = (const float*)d_in[14];
    p.eg_b1   = (const float*)d_in[15];
    p.eg_w2   = (const float*)d_in[16];
    p.eg_b2   = (const float*)d_in[17];
    p.eg_ctx_w= (const float*)d_in[18];
    p.eg_ctx_b= (const float*)d_in[19];
    p.eg_mu_w = (const float*)d_in[20];
    p.eg_mu_b = (const float*)d_in[21];
    p.eg_lv_w = (const float*)d_in[22];
    p.eg_lv_b = (const float*)d_in[23];
    p.cb_g    = (const float*)d_in[24];
    p.pr_mu_w = (const float*)d_in[25];
    p.pr_mu_b = (const float*)d_in[26];
    p.pr_lv_w = (const float*)d_in[27];
    p.pr_lv_b = (const float*)d_in[28];
    p.out     = (float*)d_out;
    p.ws      = (float*)d_ws;
    p.B       = in_sizes[0] / 64;   // 65536

    style_init<<<1, 32, 0, stream>>>(p.ws);
    style_main<<<NBLOCKS, THREADS, 0, stream>>>(p);
    style_finalize<<<1, 1, 0, stream>>>(p.ws, p.out, p.B);
}
